// WindowAttention_45707041964047
// MI455X (gfx1250) — matrix operand order repaired and checked by reading, not compiled, not run on hardware
//
#include <hip/hip_runtime.h>
#include <hip/hip_bf16.h>

// ---------------------------------------------------------------------------
// CDNA5 (gfx1250) windowed-attention block, all matmuls on v_wmma_f32_16x16x32_bf16.
// Shapes hardcoded to the reference: B=8, C=256, H=W=256, WS=8, HEADS=4, dh=64.
//
// Pipeline:
//   K0: convert qkv_w / proj_w fp32 -> bf16 (once, L2-resident thereafter)
//   K1: qkv GEMM  [768x256]@[256x65536] per batch  (double-buffered LDS stage)
//   K2: windowed MHA, 8192 windows x 4 heads       (all-LDS, 4 waves/WG)
//   K3: proj GEMM [256x256]@[256x65536] per batch  -> fp32 output
// ---------------------------------------------------------------------------

typedef __bf16 bf16;
typedef __attribute__((ext_vector_type(16))) __bf16 v16bf;
typedef __attribute__((ext_vector_type(8)))  float  v8f;

#define CIN   256
#define HWTOT 65536   // 256*256
#define NHEAD 4

__device__ __forceinline__ bf16 to_b(float x) { return (bf16)x; }
__device__ __forceinline__ bf16 to_b(bf16  x) { return x; }
__device__ __forceinline__ void store_out(float* p, float v) { *p = v; }
__device__ __forceinline__ void store_out(bf16*  p, float v) { *p = (bf16)v; }

__device__ __forceinline__ v8f zero8() {
  v8f z;
#pragma unroll
  for (int i = 0; i < 8; ++i) z[i] = 0.0f;
  return z;
}

// A fragment (16x32 bf16, MxK) from LDS tile stored row-major [m][k], leading dim ld.
// ISA layout: lane m = l%16, half = l/16; elem i -> K = (i/8)*16 + half*8 + i%8.
__device__ __forceinline__ v16bf frag_a_lds(const bf16* base, int ld) {
  int l = threadIdx.x & 31;
  int m = l & 15, h = l >> 4;
  const bf16* p = base + m * ld + h * 8;
  v16bf a;
#pragma unroll
  for (int i = 0; i < 8; ++i) a[i] = p[i];
#pragma unroll
  for (int i = 0; i < 8; ++i) a[8 + i] = p[16 + i];
  return a;
}

// B fragment (32x16 bf16, KxN) from LDS stored as B^T: rows [n][k], leading dim ld.
// ISA layout: lane n = l%16, half = l/16; elem i -> K = (i/8)*16 + half*8 + i%8.
__device__ __forceinline__ v16bf frag_b_lds(const bf16* baseT, int ld) {
  int l = threadIdx.x & 31;
  int n = l & 15, h = l >> 4;
  const bf16* p = baseT + n * ld + h * 8;
  v16bf b;
#pragma unroll
  for (int i = 0; i < 8; ++i) b[i] = p[i];
#pragma unroll
  for (int i = 0; i < 8; ++i) b[8 + i] = p[16 + i];
  return b;
}

// A fragment from a global *bf16* weight matrix W[row][c] (row-major, ldw).
// Two 16B loads per lane; weights are tiny (<=384KB) -> L2 resident.
__device__ __forceinline__ v16bf frag_a_gbf16(const bf16* W, int ldw, int row0, int k0) {
  int l = threadIdx.x & 31;
  int m = l & 15, h = l >> 4;
  const bf16* p = W + (size_t)(row0 + m) * ldw + k0 + h * 8;
  v16bf a;
#pragma unroll
  for (int i = 0; i < 8; ++i) a[i] = p[i];
#pragma unroll
  for (int i = 0; i < 8; ++i) a[8 + i] = p[16 + i];
  return a;
}

// ---------------------------------------------------------------------------
// K0: one-shot fp32 -> bf16 weight conversion (qkv_w 768x256, proj_w 256x256)
// ---------------------------------------------------------------------------
__global__ __launch_bounds__(256) void cvt_weights(
    const float* __restrict__ qw, const float* __restrict__ pw,
    bf16* __restrict__ qwb, bf16* __restrict__ pwb) {
  int i = blockIdx.x * 256 + threadIdx.x;
  if (i < 768 * 256) qwb[i] = (bf16)qw[i];
  if (i < 256 * 256) pwb[i] = (bf16)pw[i];
}

// ---------------------------------------------------------------------------
// K1/K3: 1x1 conv as GEMM:  Y[b,o,:] = W[o,:] @ X[b,:,:] + bias[o]
// Workgroup tile 128(M) x 128(N); 8 waves, each 64x32 (4 Mfrag x 2 Nfrag).
// Double-buffered LDS stage: global loads for step k+1 overlap WMMAs of step k.
// ---------------------------------------------------------------------------
template <typename InT, typename OutT>
__global__ __launch_bounds__(256) void conv1x1_wmma(
    const bf16* __restrict__ Wb, const float* __restrict__ bias,
    const InT* __restrict__ X, OutT* __restrict__ Y, int Mtot) {
  constexpr int ld = 40;  // padded leading dim (80B rows: 16B-aligned, bank-spread)
  __shared__ __align__(16) bf16 Xt[2][128 * ld];  // B^T tiles: [n][k], double-buffered

  const int n0 = blockIdx.x * 128;
  const int m0 = blockIdx.y * 128;
  const int b  = blockIdx.z;
  const int tid = threadIdx.x;
  const int wid = tid >> 5;
  const int mw = wid & 1;   // 0..1 -> M rows [mw*64, +64)
  const int nw = wid >> 1;  // 0..3 -> N cols [nw*32, +32)

  const InT* Xb = X + (size_t)b * CIN * HWTOT;
  const int cl = tid >> 3;  // 0..31 : K row of stage
  const int ch = tid & 7;   // 16-wide N chunk

  v8f acc[4][2];
#pragma unroll
  for (int i = 0; i < 4; ++i)
#pragma unroll
    for (int j = 0; j < 2; ++j) acc[i][j] = zero8();

  // prologue: stage K-tile 0 into buffer 0
  {
    const InT* src = Xb + (size_t)cl * HWTOT + n0 + ch * 16;
#pragma unroll
    for (int j = 0; j < 16; ++j) Xt[0][(ch * 16 + j) * ld + cl] = to_b(src[j]);
  }
  __syncthreads();

  for (int step = 0; step < 8; ++step) {
    const int k0 = step * 32;
    const int cur = step & 1;
    const bool more = (step < 7);

    // issue next K-tile's global loads early (latency hidden behind WMMAs)
    InT rx[16];
    if (more) {
      const InT* src = Xb + (size_t)(k0 + 32 + cl) * HWTOT + n0 + ch * 16;
#pragma unroll
      for (int j = 0; j < 16; ++j) rx[j] = src[j];
      if (step < 6)  // prefetch two K-steps ahead -> global_prefetch_b8
        __builtin_prefetch(Xb + (size_t)(k0 + 64 + cl) * HWTOT + n0 + ch * 16, 0, 0);
    }

    // compute on current buffer
    v16bf bfr[2];
#pragma unroll
    for (int nf = 0; nf < 2; ++nf)
      bfr[nf] = frag_b_lds(&Xt[cur][(nw * 32 + nf * 16) * ld], ld);
#pragma unroll
    for (int mf = 0; mf < 4; ++mf) {
      v16bf af = frag_a_gbf16(Wb, CIN, m0 + mw * 64 + mf * 16, k0);
#pragma unroll
      for (int nf = 0; nf < 2; ++nf)
        acc[mf][nf] = __builtin_amdgcn_wmma_f32_16x16x32_bf16(
            false, af, false, bfr[nf], (short)0, acc[mf][nf], false, false);
    }

    // commit staged registers into the other buffer
    if (more) {
#pragma unroll
      for (int j = 0; j < 16; ++j) Xt[1 - cur][(ch * 16 + j) * ld + cl] = to_b(rx[j]);
    }
    __syncthreads();
  }

  // epilogue: D layout lane -> [M = r + (l/16)*8][N = l%16]
  const int l = tid & 31, lm = l & 15, h = l >> 4;
#pragma unroll
  for (int mf = 0; mf < 4; ++mf)
#pragma unroll
    for (int nf = 0; nf < 2; ++nf)
#pragma unroll
      for (int r = 0; r < 8; ++r) {
        int row = m0 + mw * 64 + mf * 16 + r + h * 8;
        int col = n0 + nw * 32 + nf * 16 + lm;
        float v = acc[mf][nf][r] + bias[row];
        store_out(&Y[((size_t)b * Mtot + row) * HWTOT + col], v);
      }
}

// ---------------------------------------------------------------------------
// K2: windowed attention. One workgroup per (window, head), 128 threads.
// Head channel split per reference: head hd owns qkv channels [hd*192, hd*192+192):
//   q = +d, k = +64+d, v = +128+d  (d in [0,64)).
// ---------------------------------------------------------------------------
__global__ __launch_bounds__(128) void win_attn_wmma(
    const bf16* __restrict__ qkv, bf16* __restrict__ attn_out) {
  constexpr int ld = 72;  // 144B rows: 16B-aligned, bank-spread
  __shared__ __align__(16) bf16 Qs[64 * ld];  // [q-token][d]   (A of S; reused as Os[d][token])
  __shared__ __align__(16) bf16 Ks[64 * ld];  // [k-token][d]   (B^T of S)
  __shared__ __align__(16) bf16 Vt[64 * ld];  // [d][k-token]   (B^T of O)
  __shared__ __align__(16) bf16 Ps[64 * ld];  // [q-token][k-token] (A of O)

  const int blk  = blockIdx.x;
  const int hd   = blk & 3;
  const int widx = blk >> 2;
  const int wi = widx & 31, hi = (widx >> 5) & 31, b = widx >> 10;
  const int y0 = hi * 8, x0 = wi * 8;
  const int tid = threadIdx.x;

  const bf16* qkvb = qkv + (size_t)b * 768 * HWTOT;

  {  // stage q, k, v (each thread: one d-channel, half the tokens; 16B global loads)
    const int d = tid & 63;
    const int seg = tid >> 6;  // 0..1
    const bf16* qsrc = qkvb + (size_t)(hd * 192 + d) * HWTOT + (size_t)y0 * 256 + x0;
    const bf16* ksrc = qkvb + (size_t)(hd * 192 + 64 + d) * HWTOT + (size_t)y0 * 256 + x0;
    const bf16* vsrc = qkvb + (size_t)(hd * 192 + 128 + d) * HWTOT + (size_t)y0 * 256 + x0;
#pragma unroll
    for (int ty = 0; ty < 4; ++ty) {
      int tyy = seg * 4 + ty;
#pragma unroll
      for (int tx = 0; tx < 8; ++tx) {
        int t = tyy * 8 + tx;
        Qs[t * ld + d] = qsrc[tyy * 256 + tx];
        Ks[t * ld + d] = ksrc[tyy * 256 + tx];
        Vt[d * ld + t] = vsrc[tyy * 256 + tx];
      }
    }
  }
  __syncthreads();

  const int wv = tid >> 5;  // wave owns q-token rows [wv*16, +16)
  const int l = tid & 31, lm = l & 15, h = l >> 4;

  // S = Q @ K^T   (M = q-token, N = k-token, K = d)
  v8f s[4];
#pragma unroll
  for (int nt = 0; nt < 4; ++nt) s[nt] = zero8();
#pragma unroll
  for (int k0 = 0; k0 < 64; k0 += 32) {
    v16bf a = frag_a_lds(&Qs[(wv * 16) * ld + k0], ld);
#pragma unroll
    for (int nt = 0; nt < 4; ++nt) {
      v16bf bb = frag_b_lds(&Ks[(nt * 16) * ld + k0], ld);
      s[nt] = __builtin_amdgcn_wmma_f32_16x16x32_bf16(false, a, false, bb,
                                                      (short)0, s[nt], false, false);
    }
  }

  // softmax over full 64-wide rows: row m lives in VGPR r across 16 lanes (per half)
  const float scale = 0.125f;  // dh^-0.5
#pragma unroll
  for (int r = 0; r < 8; ++r) {
    float mx = -3.0e38f;
#pragma unroll
    for (int nt = 0; nt < 4; ++nt) mx = fmaxf(mx, s[nt][r]);
#pragma unroll
    for (int off = 8; off >= 1; off >>= 1) mx = fmaxf(mx, __shfl_xor(mx, off, 32));
    float sum = 0.0f;
#pragma unroll
    for (int nt = 0; nt < 4; ++nt) {
      float p = __expf((s[nt][r] - mx) * scale);
      s[nt][r] = p;
      sum += p;
    }
#pragma unroll
    for (int off = 8; off >= 1; off >>= 1) sum += __shfl_xor(sum, off, 32);
    float inv = 1.0f / sum;
#pragma unroll
    for (int nt = 0; nt < 4; ++nt) s[nt][r] *= inv;
  }

  // P -> LDS in A layout [q-token][k-token]
#pragma unroll
  for (int r = 0; r < 8; ++r)
#pragma unroll
    for (int nt = 0; nt < 4; ++nt)
      Ps[(wv * 16 + r + h * 8) * ld + nt * 16 + lm] = (bf16)s[nt][r];
  __syncthreads();  // Ps ready; Qs/Ks reads complete -> Qs reusable below

  // O = P @ V   (M = q-token, N = d, K = k-token)
  v8f o[4];
#pragma unroll
  for (int nt = 0; nt < 4; ++nt) o[nt] = zero8();
#pragma unroll
  for (int k0 = 0; k0 < 64; k0 += 32) {
    v16bf a = frag_a_lds(&Ps[(wv * 16) * ld + k0], ld);
#pragma unroll
    for (int nt = 0; nt < 4; ++nt) {
      v16bf bb = frag_b_lds(&Vt[(nt * 16) * ld + k0], ld);
      o[nt] = __builtin_amdgcn_wmma_f32_16x16x32_bf16(false, a, false, bb,
                                                      (short)0, o[nt], false, false);
    }
  }

  // transpose through LDS (Os[d][token]) for contiguous global stores
  bf16* Os = Qs;
#pragma unroll
  for (int r = 0; r < 8; ++r)
#pragma unroll
    for (int nt = 0; nt < 4; ++nt)
      Os[(nt * 16 + lm) * ld + wv * 16 + r + h * 8] = (bf16)o[nt][r];
  __syncthreads();

  {  // out channel = hd*64 + d (reference's head-major output layout)
    const int d = tid & 63;
    const int seg = tid >> 6;
    bf16* dst = attn_out + ((size_t)b * 256 + hd * 64 + d) * HWTOT + (size_t)y0 * 256 + x0;
#pragma unroll
    for (int ty = 0; ty < 4; ++ty) {
      int tyy = seg * 4 + ty;
#pragma unroll
      for (int tx = 0; tx < 8; ++tx) dst[tyy * 256 + tx] = Os[d * ld + tyy * 8 + tx];
    }
  }
}

// ---------------------------------------------------------------------------
extern "C" void kernel_launch(void* const* d_in, const int* in_sizes, int n_in,
                              void* d_out, int out_size, void* d_ws, size_t ws_size,
                              hipStream_t stream) {
  const float* x      = (const float*)d_in[0];
  const float* qkv_w  = (const float*)d_in[1];
  const float* qkv_b  = (const float*)d_in[2];
  const float* proj_w = (const float*)d_in[3];
  const float* proj_b = (const float*)d_in[4];
  float* out = (float*)d_out;

  // workspace layout:
  //   [0)                qkv bf16   : 8*768*65536*2 = 805,306,368 B
  //   [+805306368)       attn bf16  : 8*256*65536*2 = 268,435,456 B
  //   [+1073741824)      qkv_w bf16 : 768*256*2     = 393,216 B
  //   [+1074135040)      proj_w bf16: 256*256*2     = 131,072 B
  char* ws = (char*)d_ws;
  bf16* qkv_ws  = (bf16*)ws;
  bf16* attn_ws = (bf16*)(ws + (size_t)8 * 768 * HWTOT * sizeof(bf16));
  bf16* qw_b    = (bf16*)(ws + (size_t)1073741824);
  bf16* pw_b    = (bf16*)(ws + (size_t)1073741824 + 768 * 256 * sizeof(bf16));

  // K0: weight conversion (once per call; tiny)
  cvt_weights<<<dim3(768), dim3(256), 0, stream>>>(qkv_w, proj_w, qw_b, pw_b);

  // K1: qkv = qkv_w @ x + qkv_b            [768 x 256] @ [256 x 65536] per batch
  conv1x1_wmma<float, bf16><<<dim3(512, 6, 8), dim3(256), 0, stream>>>(
      qw_b, qkv_b, x, qkv_ws, 768);

  // K2: windowed multi-head attention      8192 windows x 4 heads
  win_attn_wmma<<<dim3(8192 * NHEAD), dim3(128), 0, stream>>>(qkv_ws, attn_ws);

  // K3: out = proj_w @ attn + proj_b       [256 x 256] @ [256 x 65536] per batch
  conv1x1_wmma<bf16, float><<<dim3(512, 2, 8), dim3(256), 0, stream>>>(
      pw_b, proj_b, attn_ws, out, 256);
}
